// LSTM_13967233647613
// MI455X (gfx1250) — matrix operand order repaired and checked by reading, not compiled, not run on hardware
//
#include <hip/hip_runtime.h>
#include <hip/hip_bf16.h>
#include <math.h>

typedef __attribute__((ext_vector_type(16))) __bf16 v16bf;
typedef __attribute__((ext_vector_type(8)))  __bf16 v8bf;
typedef __attribute__((ext_vector_type(8)))  float  v8f;

#define B_    256
#define T_    512
#define D_    256
#define H_    1024
#define O_    128
#define KTOT  1280          // D + H
#define FOURH 4096
#define LDSP  72            // 64 + 8 pad (elements)

// ---- workspace layout (bytes) ----
#define OFF_W   ((size_t)0)                         // bf16 [4096][1280]  = 10485760
#define OFF_BX  ((size_t)10485760)                  // f32  [4096]        = 16384
#define OFF_XT  ((size_t)10502144)                  // bf16 [512][256][256] = 67108864
#define OFF_H0  ((size_t)77611008)                  // bf16 [256][1024]   = 524288
#define OFF_H1  ((size_t)78135296)                  // bf16 [256][1024]   = 524288
#define OFF_CNT ((size_t)78659584)                  // u32 counter

#if defined(__AMDGCN__) && __has_builtin(__builtin_amdgcn_global_load_async_to_lds_b128)
#define USE_ASYNC_LDS 1
#else
#define USE_ASYNC_LDS 0
#endif

static __device__ __forceinline__ v16bf make16(v8bf lo, v8bf hi) {
  return __builtin_shufflevector(lo, hi, 0,1,2,3,4,5,6,7,8,9,10,11,12,13,14,15);
}
static __device__ __forceinline__ float sigm(float x) { return 1.f / (1.f + __expf(-x)); }
static __device__ __forceinline__ float tanh_f(float x) { return 2.f / (1.f + __expf(-2.f * x)) - 1.f; }

#if USE_ASYNC_LDS
// param types per hipcc diagnostic: (v4i AS1*, v4i AS3*, imm offset, imm cpol)
typedef int v4i_g __attribute__((vector_size(4 * sizeof(int))));
typedef __attribute__((address_space(1))) v4i_g* gptr128_t;
typedef __attribute__((address_space(3))) v4i_g* lptr128_t;

static __device__ __forceinline__ void async_cp16(const __bf16* g, __bf16* l) {
  __builtin_amdgcn_global_load_async_to_lds_b128((gptr128_t)(g), (lptr128_t)(l), 0, 0);
}
static __device__ __forceinline__ void wait_async0() {
#if __has_builtin(__builtin_amdgcn_s_wait_asynccnt)
  __builtin_amdgcn_s_wait_asynccnt(0);
#else
  asm volatile("s_wait_asynccnt 0x0" ::: "memory");
#endif
}
#endif

// -------------------- packing kernels --------------------
__global__ void pack_w_kernel(const float* __restrict__ Wgx, const float* __restrict__ Wgh,
                              const float* __restrict__ Wix, const float* __restrict__ Wih,
                              const float* __restrict__ Wfx, const float* __restrict__ Wfh,
                              const float* __restrict__ Wox, const float* __restrict__ Woh,
                              __bf16* __restrict__ Wcat) {
  size_t idx = (size_t)blockIdx.x * 256 + threadIdx.x;
  if (idx >= (size_t)FOURH * KTOT) return;
  int k = (int)(idx % KTOT);
  int n = (int)(idx / KTOT);
  int q = n >> 10, r = n & 1023;
  const float* Wx; const float* Wh;
  if (q == 0)      { Wx = Wgx; Wh = Wgh; }
  else if (q == 1) { Wx = Wix; Wh = Wih; }
  else if (q == 2) { Wx = Wfx; Wh = Wfh; }
  else             { Wx = Wox; Wh = Woh; }
  float v = (k < D_) ? Wx[(size_t)r * D_ + k] : Wh[(size_t)r * H_ + (k - D_)];
  Wcat[idx] = (__bf16)v;
}

__global__ void pack_bias_kernel(const float* __restrict__ bgx, const float* __restrict__ bix,
                                 const float* __restrict__ bfx, const float* __restrict__ box_,
                                 float* __restrict__ bx) {
  int idx = blockIdx.x * 256 + threadIdx.x;
  if (idx >= FOURH) return;
  int q = idx >> 10, r = idx & 1023;
  const float* b = (q == 0) ? bgx : (q == 1) ? bix : (q == 2) ? bfx : box_;
  bx[idx] = b[r];
}

__global__ void pack_x_kernel(const float* __restrict__ x, __bf16* __restrict__ xT) {
  size_t idx = (size_t)blockIdx.x * 256 + threadIdx.x;     // over T*B*D
  if (idx >= (size_t)T_ * B_ * D_) return;
  int d = (int)(idx % D_);
  size_t tb = idx / D_;
  int b = (int)(tb % B_);
  int t = (int)(tb / B_);
  xT[idx] = (__bf16)x[(size_t)b * T_ * D_ + (size_t)t * D_ + d];
}

__global__ void init_state_kernel(unsigned int* __restrict__ h0_u32, unsigned int* __restrict__ cnt) {
  int idx = blockIdx.x * 256 + threadIdx.x;
  if (idx < (B_ * H_ * 2) / 4) h0_u32[idx] = 0u;   // zero h0 (bf16 zeros)
  if (idx == 0) *cnt = 0u;
}

// -------------------- grid sync --------------------
static __device__ __forceinline__ void gsync(unsigned int* cnt, unsigned int target) {
  __threadfence();                       // release: flush this thread's h stores
  __syncthreads();
  if (threadIdx.x == 0) {
    __hip_atomic_fetch_add(cnt, 1u, __ATOMIC_RELEASE, __HIP_MEMORY_SCOPE_AGENT);
    while (__hip_atomic_load(cnt, __ATOMIC_ACQUIRE, __HIP_MEMORY_SCOPE_AGENT) < target) {
      __builtin_amdgcn_s_sleep(1);
    }
  }
  __syncthreads();
  __builtin_amdgcn_fence(__ATOMIC_ACQUIRE, "agent");   // invalidate stale h in near caches
}

// -------------------- persistent LSTM kernel --------------------
// grid = 32 WGs (each owns 32 H-columns: all 4 gates), block = 256 (8 waves).
// Per wave: M-tiles {2w, 2w+1} (rows 32w..32w+31), N-tiles j=0..7 (gate q=j>>1, sub=j&1).
__launch_bounds__(256, 1)
__global__ void lstm_kernel(const __bf16* __restrict__ Wcat,
                            const float*  __restrict__ bx,
                            const __bf16* __restrict__ xT,
                            __bf16* __restrict__ h0,
                            __bf16* __restrict__ h1,
                            unsigned int* __restrict__ cnt) {
  __shared__ __bf16 sB[2][128 * LDSP];

  const int tid  = threadIdx.x;
  const int lane = tid & 31;
  const int wv   = tid >> 5;     // 0..7
  const int bid  = blockIdx.x;   // 0..31 (H-slice)
  const int l15  = lane & 15;
  const int lhi  = lane >> 4;    // 0/1

  // per-lane input-side biases: bias[q][sub]
  float bias[4][2];
#pragma unroll
  for (int q = 0; q < 4; ++q)
#pragma unroll
    for (int sub = 0; sub < 2; ++sub)
      bias[q][sub] = bx[q * 1024 + bid * 32 + sub * 16 + l15];

  // cell state kept in registers for all T steps
  float creg[2][2][8];
#pragma unroll
  for (int mi = 0; mi < 2; ++mi)
#pragma unroll
    for (int sub = 0; sub < 2; ++sub)
#pragma unroll
      for (int r = 0; r < 8; ++r) creg[mi][sub][r] = 0.f;

  // staging assignment: thread -> (row 0..127, k-half)
  const int srow = tid >> 1;
  const int sq   = srow >> 5, sc = srow & 31;
  const int skh  = (tid & 1) * 32;
  const __bf16* wsrc = Wcat + (size_t)(sq * 1024 + bid * 32 + sc) * KTOT + skh;
  __bf16* sdst0 = &sB[0][srow * LDSP + skh];
  __bf16* sdst1 = &sB[1][srow * LDSP + skh];

  const size_t arow0 = (size_t)(wv * 32 + l15);        // global row of A, m-tile 0
  const size_t arow1 = (size_t)(wv * 32 + 16 + l15);   // m-tile 1

#pragma unroll 1
  for (int t = 0; t < T_; ++t) {
    const __bf16* hread  = (t & 1) ? h1 : h0;
    __bf16*       hwrite = (t & 1) ? h0 : h1;
    const __bf16* xrow   = xT + (size_t)t * (B_ * D_);

    v8f acc[2][8];
#pragma unroll
    for (int mi = 0; mi < 2; ++mi)
#pragma unroll
      for (int j = 0; j < 8; ++j)
#pragma unroll
        for (int r = 0; r < 8; ++r) acc[mi][j][r] = 0.f;

    // preload chunk 0 into buffer 0
#if USE_ASYNC_LDS
    async_cp16(wsrc + 0,  sdst0 + 0);
    async_cp16(wsrc + 8,  sdst0 + 8);
    async_cp16(wsrc + 16, sdst0 + 16);
    async_cp16(wsrc + 24, sdst0 + 24);
    wait_async0();
#else
    {
      uint4 s0 = *(const uint4*)(wsrc + 0);
      uint4 s1 = *(const uint4*)(wsrc + 8);
      uint4 s2 = *(const uint4*)(wsrc + 16);
      uint4 s3 = *(const uint4*)(wsrc + 24);
      uint4* d = (uint4*)sdst0;
      d[0] = s0; d[1] = s1; d[2] = s2; d[3] = s3;
    }
#endif
    __syncthreads();

#pragma unroll 1
    for (int kc = 0; kc < 20; ++kc) {
#if USE_ASYNC_LDS
      if (kc < 19) {   // async DMA next chunk straight into the other LDS buffer
        const __bf16* ns = wsrc + (size_t)(kc + 1) * 64;
        __bf16* d = (kc & 1) ? sdst0 : sdst1;
        async_cp16(ns + 0,  d + 0);
        async_cp16(ns + 8,  d + 8);
        async_cp16(ns + 16, d + 16);
        async_cp16(ns + 24, d + 24);
      }
#else
      uint4 n0, n1, n2, n3;
      if (kc < 19) {
        const __bf16* ns = wsrc + (size_t)(kc + 1) * 64;
        n0 = *(const uint4*)(ns + 0);
        n1 = *(const uint4*)(ns + 8);
        n2 = *(const uint4*)(ns + 16);
        n3 = *(const uint4*)(ns + 24);
      }
#endif
      const __bf16* lbuf = sB[kc & 1];

#pragma unroll
      for (int ks = 0; ks < 2; ++ks) {
        const int kk = kc * 64 + ks * 32;
        v16bf a0, a1;
        if (kk < D_) {
          const __bf16* p0 = xrow + arow0 * D_ + kk + lhi * 8;
          const __bf16* p1 = xrow + arow1 * D_ + kk + lhi * 8;
          a0 = make16(*(const v8bf*)p0, *(const v8bf*)(p0 + 16));
          a1 = make16(*(const v8bf*)p1, *(const v8bf*)(p1 + 16));
        } else {
          const int kh = kk - D_;
          const __bf16* p0 = hread + arow0 * H_ + kh + lhi * 8;
          const __bf16* p1 = hread + arow1 * H_ + kh + lhi * 8;
          a0 = make16(*(const v8bf*)p0, *(const v8bf*)(p0 + 16));
          a1 = make16(*(const v8bf*)p1, *(const v8bf*)(p1 + 16));
        }
        // software-pipelined B fragments (load j+1 while j's WMMAs execute)
        const __bf16* lp0 = lbuf + (size_t)l15 * LDSP + ks * 32 + lhi * 8;
        v16bf bcur = make16(*(const v8bf*)lp0, *(const v8bf*)(lp0 + 16));
#pragma unroll
        for (int j = 0; j < 8; ++j) {
          v16bf bnxt;
          if (j < 7) {
            const __bf16* lp = lbuf + (size_t)((j + 1) * 16 + l15) * LDSP + ks * 32 + lhi * 8;
            bnxt = make16(*(const v8bf*)lp, *(const v8bf*)(lp + 16));
          }
          acc[0][j] = __builtin_amdgcn_wmma_f32_16x16x32_bf16(
              false, a0, false, bcur, (short)0, acc[0][j], false, false);
          acc[1][j] = __builtin_amdgcn_wmma_f32_16x16x32_bf16(
              false, a1, false, bcur, (short)0, acc[1][j], false, false);
          if (j < 7) bcur = bnxt;
        }
      }

#if USE_ASYNC_LDS
      if (kc < 19) wait_async0();
#else
      if (kc < 19) {
        uint4* d = (uint4*)((kc & 1) ? sdst0 : sdst1);
        d[0] = n0; d[1] = n1; d[2] = n2; d[3] = n3;
      }
#endif
      __syncthreads();
    }

    // ---- gates + state update (all in registers) ----
#pragma unroll
    for (int mi = 0; mi < 2; ++mi) {
#pragma unroll
      for (int sub = 0; sub < 2; ++sub) {
        const int col = bid * 32 + sub * 16 + l15;
        v8f zg = acc[mi][0 + sub];
        v8f zi = acc[mi][2 + sub];
        v8f zf = acc[mi][4 + sub];
        v8f zo = acc[mi][6 + sub];
#pragma unroll
        for (int r = 0; r < 8; ++r) {
          float g  = tanh_f(zg[r] + bias[0][sub]);
          float ii = sigm(zi[r] + bias[1][sub]);
          float ff = sigm(zf[r] + bias[2][sub]);
          float oo = sigm(zo[r] + bias[3][sub]);
          float cv = g * ii + creg[mi][sub][r] * ff;
          creg[mi][sub][r] = cv;
          float hv = tanh_f(cv) * oo;
          const int row = (wv * 2 + mi) * 16 + r + lhi * 8;
          hwrite[(size_t)row * H_ + col] = (__bf16)hv;
        }
      }
    }

    if (t != T_ - 1) gsync(cnt, (unsigned)(gridDim.x * (t + 1)));
  }
}

// -------------------- output head: p = h @ Why^T + bhy; softmax --------------------
__global__ void head_kernel(const __bf16* __restrict__ h,
                            const float* __restrict__ Why,
                            const float* __restrict__ bhy,
                            float* __restrict__ out) {
  __shared__ float red[O_];
  const int b = blockIdx.x;
  const int o = threadIdx.x;   // 0..127
  const __bf16* hr = h + (size_t)b * H_;
  const float*  wr = Why + (size_t)o * H_;
  float acc = bhy[o];
#pragma unroll 8
  for (int k = 0; k < H_; ++k) acc += (float)hr[k] * wr[k];

  red[o] = acc; __syncthreads();
  for (int s = 64; s > 0; s >>= 1) { if (o < s) red[o] = fmaxf(red[o], red[o + s]); __syncthreads(); }
  float mx = red[0]; __syncthreads();
  float e = __expf(acc - mx);
  red[o] = e; __syncthreads();
  for (int s = 64; s > 0; s >>= 1) { if (o < s) red[o] += red[o + s]; __syncthreads(); }
  out[(size_t)b * O_ + o] = e / red[0];
}

// -------------------- launcher --------------------
extern "C" void kernel_launch(void* const* d_in, const int* in_sizes, int n_in,
                              void* d_out, int out_size, void* d_ws, size_t ws_size,
                              hipStream_t stream) {
  (void)in_sizes; (void)n_in; (void)out_size; (void)ws_size;
  const float* x    = (const float*)d_in[0];
  const float* Wgx  = (const float*)d_in[1];
  const float* bgx  = (const float*)d_in[2];
  const float* Wgh  = (const float*)d_in[3];
  const float* Wix  = (const float*)d_in[4];
  const float* bix  = (const float*)d_in[5];
  const float* Wih  = (const float*)d_in[6];
  const float* Wfx  = (const float*)d_in[7];
  const float* bfx  = (const float*)d_in[8];
  const float* Wfh  = (const float*)d_in[9];
  const float* Wox  = (const float*)d_in[10];
  const float* box_ = (const float*)d_in[11];
  const float* Woh  = (const float*)d_in[12];
  const float* Why  = (const float*)d_in[13];
  const float* bhy  = (const float*)d_in[14];

  char* ws = (char*)d_ws;
  __bf16* Wcat      = (__bf16*)(ws + OFF_W);
  float*  bx        = (float*)(ws + OFF_BX);
  __bf16* xT        = (__bf16*)(ws + OFF_XT);
  __bf16* h0        = (__bf16*)(ws + OFF_H0);
  __bf16* h1        = (__bf16*)(ws + OFF_H1);
  unsigned int* cnt = (unsigned int*)(ws + OFF_CNT);

  pack_w_kernel<<<(FOURH * KTOT + 255) / 256, 256, 0, stream>>>(
      Wgx, Wgh, Wix, Wih, Wfx, Wfh, Wox, Woh, Wcat);
  pack_bias_kernel<<<(FOURH + 255) / 256, 256, 0, stream>>>(bgx, bix, bfx, box_, bx);
  pack_x_kernel<<<(T_ * B_ * D_ + 255) / 256, 256, 0, stream>>>(x, xT);
  init_state_kernel<<<(B_ * H_ * 2 / 4 + 255) / 256, 256, 0, stream>>>(
      (unsigned int*)h0, cnt);

  lstm_kernel<<<32, 256, 0, stream>>>(Wcat, bx, xT, h0, h1, cnt);

  // T=512 is even -> final h lives in h0
  head_kernel<<<B_, O_, 0, stream>>>(h0, Why, bhy, (float*)d_out);
}
